// ResidualMambaBlock_39951785787665
// MI455X (gfx1250) — matrix operand order repaired and checked by reading, not compiled, not run on hardware
//
#include <hip/hip_runtime.h>
#include <hip/hip_bf16.h>

// ---------------- problem constants (from reference) ----------------
#define D_MODEL   1024
#define D_STATE   16
#define D_CONV    4
#define DT_RANK   64
#define D_INNER   2048            // EXPAND * D_MODEL
#define B_SZ      2
#define SEQ       4096
#define NROWS     (B_SZ * SEQ)    // 8192 token rows
#define XPROJ_N   (DT_RANK + 2 * D_STATE)  // 96
#define CS        256             // scan chunk length
#define NCH       (SEQ / CS)      // 16 chunks per sequence

// ---------------- vector types for WMMA ----------------
typedef __attribute__((ext_vector_type(16))) __bf16 v16bf;
typedef __attribute__((ext_vector_type(8)))  __bf16 v8bf;
typedef __attribute__((ext_vector_type(8)))  float  v8f;

__device__ __forceinline__ __bf16 f2bf(float f) {
  union { float f; unsigned u; } a; a.f = f;
  unsigned r = a.u + 0x7FFFu + ((a.u >> 16) & 1u);   // round-to-nearest-even
  union { unsigned short s; __bf16 b; } o;
  o.s = (unsigned short)(r >> 16);
  return o.b;
}

__device__ __forceinline__ float silu_f(float v) {
  return v / (1.0f + __expf(-v));
}

// ---------------- f32 -> bf16 convert (grid-stride) ----------------
__global__ void cvt_f32_bf16_kernel(const float* __restrict__ in,
                                    __bf16* __restrict__ out, int n) {
  for (int i = blockIdx.x * blockDim.x + threadIdx.x; i < n;
       i += gridDim.x * blockDim.x)
    out[i] = f2bf(in[i]);
}

// ---------------- LayerNorm -> bf16 activations ----------------
__global__ void ln_kernel(const float* __restrict__ x,
                          const float* __restrict__ w,
                          const float* __restrict__ b,
                          __bf16* __restrict__ h) {
  __shared__ float ssum[256], ssq[256];
  const int row = blockIdx.x;
  const float* xr = x + (size_t)row * D_MODEL;
  float s = 0.f, q = 0.f;
  for (int i = threadIdx.x; i < D_MODEL; i += 256) {
    float v = xr[i]; s += v; q += v * v;
  }
  ssum[threadIdx.x] = s; ssq[threadIdx.x] = q;
  __syncthreads();
  for (int o = 128; o > 0; o >>= 1) {
    if (threadIdx.x < o) {
      ssum[threadIdx.x] += ssum[threadIdx.x + o];
      ssq[threadIdx.x]  += ssq[threadIdx.x + o];
    }
    __syncthreads();
  }
  const float mean = ssum[0] * (1.0f / D_MODEL);
  const float var  = ssq[0] * (1.0f / D_MODEL) - mean * mean;
  const float rstd = rsqrtf(var + 1e-5f);
  for (int i = threadIdx.x; i < D_MODEL; i += 256) {
    float v = (xr[i] - mean) * rstd * w[i] + b[i];
    h[(size_t)row * D_MODEL + i] = f2bf(v);
  }
}

// ---------------- register-blocked, software-pipelined WMMA bf16 GEMM --------
// C[M,N] = A[M,K] * W[N,K]^T  (+resid).  One wave computes a 16 x (16*TN)
// output slab; one A fragment per K-step is reused across TN B fragments.
// K-loop is software-pipelined: all global loads for step kb+32 are issued
// BEFORE the TN WMMAs of step kb, so load latency overlaps matrix-pipe work
// and the WMMA->WMMA hazard slots are filled with loads/address VALU.
// The 8 waves of a block take 8 adjacent M strips of the SAME N block, so
// their B fragments are identical -> served from L0.
template <int TN>
__global__ void gemm_bf16_wmma_kernel(const __bf16* __restrict__ A,
                                      const __bf16* __restrict__ W,
                                      float* __restrict__ C,
                                      const float* __restrict__ resid,
                                      int M, int N, int K) {
  const int lane = threadIdx.x & 31;
  const int half = lane >> 4;            // 0/1
  const int l16  = lane & 15;
  const int wid  = (blockIdx.x * blockDim.x + threadIdx.x) >> 5; // global wave
  const int tilesM = M >> 4;
  const int mb = wid % tilesM;           // consecutive waves -> adjacent M strips
  const int nb = wid / tilesM;           // shared N block within a block of waves
  const int nbase = nb * (16 * TN);

  const __bf16* arow = A + (size_t)(mb * 16 + l16) * K;
  const __bf16* bp[TN];
#pragma unroll
  for (int i = 0; i < TN; ++i)
    bp[i] = W + (size_t)(nbase + i * 16 + l16) * K;   // W row n == B column n

  v8f acc[TN];
#pragma unroll
  for (int i = 0; i < TN; ++i) acc[i] = (v8f){};

  union F16 { v16bf v; v8bf h[2]; };

  // prologue: load fragments for kb = 0
  F16 af, bfr[TN];
  af.h[0] = *(const v8bf*)(arow + half * 8);
  af.h[1] = *(const v8bf*)(arow + 16 + half * 8);
#pragma unroll
  for (int i = 0; i < TN; ++i) {
    bfr[i].h[0] = *(const v8bf*)(bp[i] + half * 8);
    bfr[i].h[1] = *(const v8bf*)(bp[i] + 16 + half * 8);
  }

  // steady state: issue kb+32 loads, then kb WMMAs
  int kb = 0;
  for (; kb + 32 < K; kb += 32) {
    F16 afn, bfn[TN];
    const int kn = kb + 32;
    afn.h[0] = *(const v8bf*)(arow + kn + half * 8);
    afn.h[1] = *(const v8bf*)(arow + kn + 16 + half * 8);
#pragma unroll
    for (int i = 0; i < TN; ++i) {
      bfn[i].h[0] = *(const v8bf*)(bp[i] + kn + half * 8);
      bfn[i].h[1] = *(const v8bf*)(bp[i] + kn + 16 + half * 8);
    }
    if (kn + 32 < K) {                   // slab after next -> global_prefetch_b8
      __builtin_prefetch((const void*)(arow + kn + 32), 0, 3);
      __builtin_prefetch((const void*)(bp[0] + kn + 32), 0, 3);
    }
#pragma unroll
    for (int i = 0; i < TN; ++i)
      acc[i] = __builtin_amdgcn_wmma_f32_16x16x32_bf16(
          false, af.v, false, bfr[i].v, (short)0, acc[i], false, false);
    af = afn;
#pragma unroll
    for (int i = 0; i < TN; ++i) bfr[i] = bfn[i];
  }
  // epilogue: last K-step
#pragma unroll
  for (int i = 0; i < TN; ++i)
    acc[i] = __builtin_amdgcn_wmma_f32_16x16x32_bf16(
        false, af.v, false, bfr[i].v, (short)0, acc[i], false, false);

  // C/D layout: VGPR r -> row (r + half*8), col l16
#pragma unroll
  for (int i = 0; i < TN; ++i) {
    const int ncol = nbase + i * 16 + l16;
#pragma unroll
    for (int r = 0; r < 8; ++r) {
      const int mrow = mb * 16 + r + half * 8;
      const size_t idx = (size_t)mrow * N + ncol;
      float v = acc[i][r];
      if (resid) v += resid[idx];
      C[idx] = v;
    }
  }
}

// ---------------- causal depthwise conv (d_conv=4) + SiLU ----------------
// xz layout: (NROWS, 2*D_INNER); xpart = cols [0, D_INNER)
__global__ void conv_silu_kernel(const float* __restrict__ xz,
                                 const float* __restrict__ cw,   // (D_INNER,1,4)
                                 const float* __restrict__ cb,
                                 float* __restrict__ u,
                                 __bf16* __restrict__ ubf) {
  const int id = blockIdx.x * blockDim.x + threadIdx.x;  // over NROWS*D_INNER
  const int c = id & (D_INNER - 1);
  const int t = (id >> 11) & (SEQ - 1);                  // D_INNER == 2048
  const int b = id >> 23;                                // / (SEQ*D_INNER)
  float acc = cb[c];
#pragma unroll
  for (int j = 0; j < D_CONV; ++j) {
    const int tl = t - (D_CONV - 1) + j;
    if (tl >= 0)
      acc += cw[c * D_CONV + j] *
             xz[((size_t)(b * SEQ + tl)) * (2 * D_INNER) + c];
  }
  const float v = silu_f(acc);
  u[(size_t)id]   = v;
  ubf[(size_t)id] = f2bf(v);
}

// ---------------- extract dt_raw (first DT_RANK cols of x_dbl) as bf16 -------
__global__ void dtraw_kernel(const float* __restrict__ xdbl,
                             __bf16* __restrict__ dtraw) {
  const int i = blockIdx.x * blockDim.x + threadIdx.x;   // over NROWS*DT_RANK
  const int row = i >> 6;                                // / 64
  const int col = i & 63;
  dtraw[i] = f2bf(xdbl[(size_t)row * XPROJ_N + col]);
}

// ---------------- softplus(lin + bias) in place ----------------
__global__ void softplus_kernel(float* __restrict__ dt,
                                const float* __restrict__ bias) {
  const int i = blockIdx.x * blockDim.x + threadIdx.x;   // over NROWS*D_INNER
  const int c = i & (D_INNER - 1);
  const float v = dt[i] + bias[c];
  dt[i] = (v > 20.f) ? v : __logf(1.0f + __expf(v));
}

// ======================= chunked selective scan ============================
// Linear recurrence h_t = exp(dt_t*A)*h_{t-1} + dt_t*u_t*B_t.  The chunk
// decay product collapses: prod_t exp(dt_t*A[n]) = exp(A[n] * sum_t dt_t).
// Pass 1: per-chunk final state (zero init) + per-chunk sum(dt).
// Pass 2: sequential chunk-boundary propagation (NCH steps per channel).
// Pass 3: re-scan each chunk from its true start state, emit gated bf16 y.
// Thread id encodes (b, j, c):  id = (b<<15) | (j<<11) | c.

__device__ __forceinline__ void load_vec16(const float* p, float* v) {
  const float4* q = (const float4*)p;
#pragma unroll
  for (int i = 0; i < 4; ++i) {
    const float4 t = q[i];
    v[4 * i + 0] = t.x; v[4 * i + 1] = t.y; v[4 * i + 2] = t.z; v[4 * i + 3] = t.w;
  }
}

__global__ void scan_pass1_kernel(const float* __restrict__ dtarr,
                                  const float* __restrict__ uarr,
                                  const float* __restrict__ xdbl,
                                  const float* __restrict__ A_log,
                                  float* __restrict__ Fbuf,   // (65536,16)
                                  float* __restrict__ Sbuf) { // (65536)
  const int id = blockIdx.x * blockDim.x + threadIdx.x;
  const int c = id & (D_INNER - 1);
  const int j = (id >> 11) & (NCH - 1);
  const int b = id >> 15;

  float Arow[D_STATE], st[D_STATE];
#pragma unroll
  for (int n = 0; n < D_STATE; ++n) {
    Arow[n] = -__expf(A_log[c * D_STATE + n]);
    st[n] = 0.f;
  }
  float S = 0.f;
  for (int t = 0; t < CS; ++t) {
    const size_t row = (size_t)(b * SEQ + j * CS + t);
    const float dt = dtarr[row * D_INNER + c];
    const float uu = uarr[row * D_INNER + c];
    float Bv[D_STATE];
    load_vec16(xdbl + row * XPROJ_N + DT_RANK, Bv);
    S += dt;
    const float du = dt * uu;
#pragma unroll
    for (int n = 0; n < D_STATE; ++n)
      st[n] = st[n] * __expf(dt * Arow[n]) + du * Bv[n];
  }
  float* F = Fbuf + (size_t)id * D_STATE;
#pragma unroll
  for (int n = 0; n < D_STATE; ++n) F[n] = st[n];
  Sbuf[id] = S;
}

__global__ void scan_pass2_kernel(const float* __restrict__ A_log,
                                  const float* __restrict__ Fbuf,
                                  const float* __restrict__ Sbuf,
                                  float* __restrict__ Hst) {  // (65536,16)
  const int tid = blockIdx.x * blockDim.x + threadIdx.x;      // over B_SZ*D_INNER
  const int c = tid & (D_INNER - 1);
  const int b = tid >> 11;

  float Arow[D_STATE], h[D_STATE];
#pragma unroll
  for (int n = 0; n < D_STATE; ++n) {
    Arow[n] = -__expf(A_log[c * D_STATE + n]);
    h[n] = 0.f;
  }
  for (int j = 0; j < NCH; ++j) {
    const int id = (b << 15) | (j << 11) | c;
    float* hs = Hst + (size_t)id * D_STATE;
    const float* F = Fbuf + (size_t)id * D_STATE;
    const float S = Sbuf[id];
#pragma unroll
    for (int n = 0; n < D_STATE; ++n) {
      hs[n] = h[n];                               // chunk-start state
      h[n] = h[n] * __expf(Arow[n] * S) + F[n];   // propagate to next chunk
    }
  }
}

__global__ void scan_pass3_kernel(const float* __restrict__ dtarr,
                                  const float* __restrict__ uarr,
                                  const float* __restrict__ xdbl,
                                  const float* __restrict__ xz,
                                  const float* __restrict__ A_log,
                                  const float* __restrict__ Dvec,
                                  const float* __restrict__ Hst,
                                  __bf16* __restrict__ ybf) {
  const int id = blockIdx.x * blockDim.x + threadIdx.x;
  const int c = id & (D_INNER - 1);
  const int j = (id >> 11) & (NCH - 1);
  const int b = id >> 15;

  float Arow[D_STATE], st[D_STATE];
  const float* hs = Hst + (size_t)id * D_STATE;
#pragma unroll
  for (int n = 0; n < D_STATE; ++n) {
    Arow[n] = -__expf(A_log[c * D_STATE + n]);
    st[n] = hs[n];
  }
  const float Dc = Dvec[c];

  for (int t = 0; t < CS; ++t) {
    const size_t row = (size_t)(b * SEQ + j * CS + t);
    const float dt = dtarr[row * D_INNER + c];
    const float uu = uarr[row * D_INNER + c];
    float Bv[D_STATE], Cv[D_STATE];
    load_vec16(xdbl + row * XPROJ_N + DT_RANK, Bv);
    load_vec16(xdbl + row * XPROJ_N + DT_RANK + D_STATE, Cv);
    const float du = dt * uu;
    float y = Dc * uu;
#pragma unroll
    for (int n = 0; n < D_STATE; ++n) {
      st[n] = st[n] * __expf(dt * Arow[n]) + du * Bv[n];
      y += st[n] * Cv[n];
    }
    const float z = xz[row * (2 * D_INNER) + D_INNER + c];
    ybf[row * D_INNER + c] = f2bf(y * silu_f(z));
  }
}

// ---------------- host-side launch ----------------
extern "C" void kernel_launch(void* const* d_in, const int* in_sizes, int n_in,
                              void* d_out, int out_size, void* d_ws, size_t ws_size,
                              hipStream_t stream) {
  const float* x        = (const float*)d_in[0];
  const float* ln_w     = (const float*)d_in[1];
  const float* ln_b     = (const float*)d_in[2];
  const float* in_pw    = (const float*)d_in[3];   // (4096,1024)
  const float* conv_w   = (const float*)d_in[4];   // (2048,1,4)
  const float* conv_b   = (const float*)d_in[5];
  const float* x_pw     = (const float*)d_in[6];   // (96,2048)
  const float* dt_pw    = (const float*)d_in[7];   // (2048,64)
  const float* dt_pb    = (const float*)d_in[8];
  const float* A_log    = (const float*)d_in[9];   // (2048,16)
  const float* Dvec     = (const float*)d_in[10];
  const float* out_pw   = (const float*)d_in[11];  // (1024,2048)
  float* out = (float*)d_out;

  char* ws = (char*)d_ws;
  const size_t OFF_HBF   = 0;                                   // 8192x1024 bf16
  const size_t OFF_WIN   = OFF_HBF   + (size_t)NROWS * D_MODEL * 2;
  const size_t OFF_WXP   = OFF_WIN   + (size_t)2 * D_INNER * D_MODEL * 2;
  const size_t OFF_WDT   = OFF_WXP   + (size_t)XPROJ_N * D_INNER * 2;
  const size_t OFF_WOUT  = OFF_WDT   + (size_t)D_INNER * DT_RANK * 2;
  const size_t OFF_XZ    = OFF_WOUT  + (size_t)D_MODEL * D_INNER * 2;
  const size_t OFF_U     = OFF_XZ    + (size_t)NROWS * 2 * D_INNER * 4;
  const size_t OFF_UBF   = OFF_U     + (size_t)NROWS * D_INNER * 4;
  const size_t OFF_XDBL  = OFF_UBF   + (size_t)NROWS * D_INNER * 2;
  const size_t OFF_DTRAW = OFF_XDBL  + (size_t)NROWS * XPROJ_N * 4;
  const size_t OFF_DT    = OFF_DTRAW + (size_t)NROWS * DT_RANK * 2;
  const size_t OFF_YBF   = OFF_DT    + (size_t)NROWS * D_INNER * 4;
  const size_t OFF_F     = OFF_YBF   + (size_t)NROWS * D_INNER * 2;
  const size_t OFF_S     = OFF_F     + (size_t)B_SZ * D_INNER * NCH * D_STATE * 4;
  const size_t OFF_HST   = OFF_S     + (size_t)B_SZ * D_INNER * NCH * 4;

  __bf16* h_bf   = (__bf16*)(ws + OFF_HBF);
  __bf16* w_in   = (__bf16*)(ws + OFF_WIN);
  __bf16* w_xp   = (__bf16*)(ws + OFF_WXP);
  __bf16* w_dt   = (__bf16*)(ws + OFF_WDT);
  __bf16* w_out  = (__bf16*)(ws + OFF_WOUT);
  float*  xz     = (float*) (ws + OFF_XZ);
  float*  u_f    = (float*) (ws + OFF_U);
  __bf16* u_bf   = (__bf16*)(ws + OFF_UBF);
  float*  xdbl   = (float*) (ws + OFF_XDBL);
  __bf16* dtraw  = (__bf16*)(ws + OFF_DTRAW);
  float*  dt_f   = (float*) (ws + OFF_DT);
  __bf16* y_bf   = (__bf16*)(ws + OFF_YBF);
  float*  Fbuf   = (float*) (ws + OFF_F);
  float*  Sbuf   = (float*) (ws + OFF_S);
  float*  Hst    = (float*) (ws + OFF_HST);

  // 1) weights -> bf16
  {
    int n1 = 2 * D_INNER * D_MODEL;
    cvt_f32_bf16_kernel<<<(n1 + 255) / 256, 256, 0, stream>>>(in_pw, w_in, n1);
    int n2 = XPROJ_N * D_INNER;
    cvt_f32_bf16_kernel<<<(n2 + 255) / 256, 256, 0, stream>>>(x_pw, w_xp, n2);
    int n3 = D_INNER * DT_RANK;
    cvt_f32_bf16_kernel<<<(n3 + 255) / 256, 256, 0, stream>>>(dt_pw, w_dt, n3);
    int n4 = D_MODEL * D_INNER;
    cvt_f32_bf16_kernel<<<(n4 + 255) / 256, 256, 0, stream>>>(out_pw, w_out, n4);
  }

  // 2) LayerNorm -> bf16 activations
  ln_kernel<<<NROWS, 256, 0, stream>>>(x, ln_w, ln_b, h_bf);

  const int tilesM = NROWS / 16;  // 512

  // 3) in_proj GEMM: (8192x1024) x (4096x1024)^T -> xz (8192x4096), TN=4
  gemm_bf16_wmma_kernel<4><<<tilesM * ((2 * D_INNER) / 64) * 32 / 256, 256, 0, stream>>>(
      h_bf, w_in, xz, nullptr, NROWS, 2 * D_INNER, D_MODEL);

  // 4) causal depthwise conv + SiLU -> u (f32 + bf16)
  conv_silu_kernel<<<(NROWS * D_INNER) / 256, 256, 0, stream>>>(
      xz, conv_w, conv_b, u_f, u_bf);

  // 5) x_proj GEMM: (8192x2048) x (96x2048)^T -> x_dbl (8192x96), TN=2 (96=3*32)
  gemm_bf16_wmma_kernel<2><<<tilesM * (XPROJ_N / 32) * 32 / 256, 256, 0, stream>>>(
      u_bf, w_xp, xdbl, nullptr, NROWS, XPROJ_N, D_INNER);

  // 6) dt_raw slice -> bf16
  dtraw_kernel<<<(NROWS * DT_RANK) / 256, 256, 0, stream>>>(xdbl, dtraw);

  // 7) dt_proj GEMM: (8192x64) x (2048x64)^T -> dt_lin (8192x2048), TN=4
  gemm_bf16_wmma_kernel<4><<<tilesM * (D_INNER / 64) * 32 / 256, 256, 0, stream>>>(
      dtraw, w_dt, dt_f, nullptr, NROWS, D_INNER, DT_RANK);

  // 8) dt = softplus(dt_lin + dt_bias), in place
  softplus_kernel<<<(NROWS * D_INNER) / 256, 256, 0, stream>>>(dt_f, dt_pb);

  // 9) chunked selective scan (3 passes) -> gated y (bf16)
  scan_pass1_kernel<<<(B_SZ * D_INNER * NCH) / 256, 256, 0, stream>>>(
      dt_f, u_f, xdbl, A_log, Fbuf, Sbuf);
  scan_pass2_kernel<<<(B_SZ * D_INNER) / 256, 256, 0, stream>>>(
      A_log, Fbuf, Sbuf, Hst);
  scan_pass3_kernel<<<(B_SZ * D_INNER * NCH) / 256, 256, 0, stream>>>(
      dt_f, u_f, xdbl, xz, A_log, Dvec, Hst, y_bf);

  // 10) out_proj GEMM + residual: out = x + (8192x2048) x (1024x2048)^T, TN=4
  gemm_bf16_wmma_kernel<4><<<tilesM * (D_MODEL / 64) * 32 / 256, 256, 0, stream>>>(
      y_bf, w_out, out, x, NROWS, D_MODEL, D_INNER);
}
